// Measure_1915555414392
// MI455X (gfx1250) — compile-verified, matrix operand order checked by reading, not run on hardware
//
#include <hip/hip_runtime.h>

typedef __attribute__((ext_vector_type(16))) __bf16 v16bf;
typedef __attribute__((ext_vector_type(8)))  float  v8f;

#define NE 16      // experts
#define ND 196     // in features
#define NH 512     // hidden
#define NA 1024    // answers
#define NB 1024    // batch
#define KDP 224    // ND padded to multiple of 32
#define KC1 7      // K-chunks for layer 1 (224/32)
#define KC2 16     // K-chunks for layer 2 (512/32)
#define NT1 32     // N-tiles layer 1 (512/16)
#define NT2 64     // N-tiles layer 2 (1024/16)
#define TPE 64     // tiles per expert = ceil(NB/16)

// d_ws layout
#define LISTS_OFF   64
#define BW1_OFF     (128 * 1024)
#define BW1_BYTES   ((size_t)NE * KC1 * NT1 * 32 * 32)        // 3,670,016
#define BW2_OFF     (BW1_OFF + 3670016)                       // 3,801,088 (32B aligned)
#define BW2_BYTES   ((size_t)NE * KC2 * NT2 * 32 * 32)        // 16,777,216
#define WS_NEEDED   ((size_t)BW2_OFF + BW2_BYTES)

__global__ void zero_counts_kernel(int* counts) {
    if (threadIdx.x < NE) counts[threadIdx.x] = 0;
}

__global__ void bucket_kernel(const int* __restrict__ instance,
                              int* __restrict__ counts,
                              int* __restrict__ lists) {
    int b = blockIdx.x * blockDim.x + threadIdx.x;
    if (b < NB) {
        int e = instance[b];
        int p = atomicAdd(&counts[e], 1);
        lists[e * NB + p] = b;
    }
}

// Convert f32 weights [E][Kvalid][N] into bf16 WMMA B-fragment order:
// dst[((e*KC + kc)*NT + ntile)*32 + lane] = v16bf fragment (ISA 7.12.2 layout).
__global__ __launch_bounds__(256) void swizzle_w_kernel(
    const float* __restrict__ W, v16bf* __restrict__ dst,
    int Kvalid, int KC, int NT, int N, int total)
{
    int id = blockIdx.x * blockDim.x + threadIdx.x;
    if (id >= total) return;
    int lane = id & 31;
    int half = lane >> 4;
    int t = id >> 5;
    int ntile = t % NT; t /= NT;
    int kc = t % KC;
    int e = t / KC;
    int n = ntile * 16 + (lane & 15);
    const float* We = W + (size_t)e * Kvalid * N;
    v16bf frag;
#pragma unroll
    for (int r = 0; r < 8; ++r) {
        int k = kc * 32 + (r & 3) * 2 + (r >> 2) * 16 + half * 8;
        frag[2 * r]     = (k     < Kvalid) ? (__bf16)We[(size_t)k       * N + n] : (__bf16)0.0f;
        frag[2 * r + 1] = (k + 1 < Kvalid) ? (__bf16)We[(size_t)(k + 1) * N + n] : (__bf16)0.0f;
    }
    dst[id] = frag;
}

// ---------------- fast path: pre-swizzled bf16 weights ----------------
__global__ __launch_bounds__(256) void moe_mlp_swz_kernel(
    const float* __restrict__ mask,
    const int*   __restrict__ counts,
    const int*   __restrict__ lists,
    const v16bf* __restrict__ BW1, const float* __restrict__ b1,
    const v16bf* __restrict__ BW2, const float* __restrict__ b2,
    float* __restrict__ out)
{
    __shared__ v16bf Xfrag[KC1 * 32];                         // 7 KB, A-frag order
    __shared__ __align__(32) __bf16 Hfrag[KC2 * 32 * 16];     // 16 KB, A-frag order
    __shared__ int sidx[16];
    __shared__ int svalid[16];

    const int e = blockIdx.x >> 6;
    const int t = blockIdx.x & 63;
    const int cnt = counts[e];
    if (t * 16 >= cnt) return;        // block-uniform: EXEC stays all-ones for WMMA

    const int tid = threadIdx.x;
    if (tid < 16) {
        int pos = t * 16 + tid;
        int v = pos < cnt;
        svalid[tid] = v;
        sidx[tid]   = v ? lists[e * NB + pos] : 0;
    }
    __syncthreads();

    // Stage A: gather X directly into A-fragment layout (one v16bf per thread)
    if (tid < KC1 * 32) {
        int kc   = tid >> 5;
        int la   = tid & 31;
        int hf   = la >> 4;
        int m    = la & 15;
        int b    = sidx[m];
        int v    = svalid[m];
        v16bf f;
#pragma unroll
        for (int r = 0; r < 8; ++r) {
            int k = kc * 32 + (r & 3) * 2 + (r >> 2) * 16 + hf * 8;
            f[2 * r]     = (v && k     < ND) ? (__bf16)mask[(size_t)b * ND + k]     : (__bf16)0.0f;
            f[2 * r + 1] = (v && k + 1 < ND) ? (__bf16)mask[(size_t)b * ND + k + 1] : (__bf16)0.0f;
        }
        Xfrag[tid] = f;
    }
    __syncthreads();

    const int lane = tid & 31;
    const int wave = tid >> 5;
    const int half = lane >> 4;
    const int ln   = lane & 15;

    // ---- Stage B: H = relu(X @ W1 + b1), each wave owns 4 N-tiles ----
    v16bf xf[KC1];
#pragma unroll
    for (int kc = 0; kc < KC1; ++kc) xf[kc] = Xfrag[kc * 32 + lane];

    {
        v8f acc[4] = {};
        for (int kc = 0; kc < KC1; ++kc) {
#pragma unroll
            for (int tt = 0; tt < 4; ++tt) {
                int nt = wave * 4 + tt;
                v16bf bf = BW1[((size_t)(e * KC1 + kc) * NT1 + nt) * 32 + lane];
                acc[tt] = __builtin_amdgcn_wmma_f32_16x16x32_bf16(
                    false, xf[kc], false, bf, (short)0, acc[tt], false, false);
            }
        }
#pragma unroll
        for (int tt = 0; tt < 4; ++tt) {
            int n = (wave * 4 + tt) * 16 + ln;
            float bias = b1[n];
            int kcc  = n >> 5;
            int ko   = n & 31;
            int elem = ((ko >> 4) & 1) * 8 + (ko & 7);   // index inside v16bf
            int hc   = (ko >> 3) & 1;                    // which lane half holds this k
#pragma unroll
            for (int r = 0; r < 8; ++r) {
                int m = r + half * 8;
                float h = acc[tt][r] + bias;
                h = h > 0.0f ? h : 0.0f;
                Hfrag[((kcc * 32) + m + 16 * hc) * 16 + elem] = (__bf16)h;
            }
        }
    }
    __syncthreads();

    // ---- Stage C: out = H @ W2 + b2, each wave owns 8 N-tiles ----
    {
        v8f acc[8] = {};
        for (int kc = 0; kc < KC2; ++kc) {
            v16bf hf = *(const v16bf*)(Hfrag + (kc * 32 + lane) * 16);
#pragma unroll
            for (int tt = 0; tt < 8; ++tt) {
                int nt = wave * 8 + tt;
                v16bf bf = BW2[((size_t)(e * KC2 + kc) * NT2 + nt) * 32 + lane];
                acc[tt] = __builtin_amdgcn_wmma_f32_16x16x32_bf16(
                    false, hf, false, bf, (short)0, acc[tt], false, false);
            }
        }
#pragma unroll
        for (int tt = 0; tt < 8; ++tt) {
            int n = (wave * 8 + tt) * 16 + ln;
            float bias = b2[n];
#pragma unroll
            for (int r = 0; r < 8; ++r) {
                int m = r + half * 8;
                if (svalid[m]) out[(size_t)sidx[m] * NA + n] = acc[tt][r] + bias;
            }
        }
    }
}

// ---------------- fallback path: direct f32 loads (small d_ws) ----------------
__global__ __launch_bounds__(256) void moe_mlp_direct_kernel(
    const float* __restrict__ mask,
    const int*   __restrict__ counts,
    const int*   __restrict__ lists,
    const float* __restrict__ W1, const float* __restrict__ b1,
    const float* __restrict__ W2, const float* __restrict__ b2,
    float* __restrict__ out)
{
    __shared__ __bf16 Xt[16 * KDP];
    __shared__ __bf16 Ht[16 * NH];
    __shared__ int    sidx[16];
    __shared__ int    svalid[16];

    const int e = blockIdx.x >> 6;
    const int t = blockIdx.x & 63;
    const int cnt = counts[e];
    if (t * 16 >= cnt) return;

    const int tid = threadIdx.x;
    if (tid < 16) {
        int pos = t * 16 + tid;
        int v = pos < cnt;
        svalid[tid] = v;
        sidx[tid]   = v ? lists[e * NB + pos] : 0;
    }
    __syncthreads();

    {
        int row = tid >> 4;
        int col = tid & 15;
        int b   = sidx[row];
        int v   = svalid[row];
        for (int k = col; k < KDP; k += 16) {
            float x = (v && k < ND) ? mask[(size_t)b * ND + k] : 0.0f;
            Xt[row * KDP + k] = (__bf16)x;
        }
    }
    __syncthreads();

    const int lane = tid & 31;
    const int wave = tid >> 5;
    const int half = lane >> 4;
    const int ln   = lane & 15;

    const float* W1e = W1 + (size_t)e * ND * NH;
    const float* W2e = W2 + (size_t)e * NH * NA;

    for (int tt = 0; tt < 4; ++tt) {
        int n = (wave * 4 + tt) * 16 + ln;
        v8f acc = {};
        for (int kc = 0; kc < KC1; ++kc) {
            v16bf afrag, bfrag;
#pragma unroll
            for (int r = 0; r < 8; ++r) {
                int k = kc * 32 + (r & 3) * 2 + (r >> 2) * 16 + half * 8;
                afrag[2 * r]     = Xt[ln * KDP + k];
                afrag[2 * r + 1] = Xt[ln * KDP + k + 1];
                bfrag[2 * r]     = (k     < ND) ? (__bf16)W1e[(size_t)k       * NH + n] : (__bf16)0.0f;
                bfrag[2 * r + 1] = (k + 1 < ND) ? (__bf16)W1e[(size_t)(k + 1) * NH + n] : (__bf16)0.0f;
            }
            acc = __builtin_amdgcn_wmma_f32_16x16x32_bf16(
                false, afrag, false, bfrag, (short)0, acc, false, false);
        }
        float bias = b1[n];
#pragma unroll
        for (int r = 0; r < 8; ++r) {
            int m = r + half * 8;
            float h = acc[r] + bias;
            h = h > 0.0f ? h : 0.0f;
            Ht[m * NH + n] = (__bf16)h;
        }
    }
    __syncthreads();

    for (int tt = 0; tt < 8; ++tt) {
        int n = (wave * 8 + tt) * 16 + ln;
        v8f acc = {};
        for (int kc = 0; kc < KC2; ++kc) {
            v16bf afrag, bfrag;
#pragma unroll
            for (int r = 0; r < 8; ++r) {
                int k = kc * 32 + (r & 3) * 2 + (r >> 2) * 16 + half * 8;
                afrag[2 * r]     = Ht[ln * NH + k];
                afrag[2 * r + 1] = Ht[ln * NH + k + 1];
                bfrag[2 * r]     = (__bf16)W2e[(size_t)k       * NA + n];
                bfrag[2 * r + 1] = (__bf16)W2e[(size_t)(k + 1) * NA + n];
            }
            acc = __builtin_amdgcn_wmma_f32_16x16x32_bf16(
                false, afrag, false, bfrag, (short)0, acc, false, false);
        }
        float bias = b2[n];
#pragma unroll
        for (int r = 0; r < 8; ++r) {
            int m = r + half * 8;
            if (svalid[m]) out[(size_t)sidx[m] * NA + n] = acc[r] + bias;
        }
    }
}

extern "C" void kernel_launch(void* const* d_in, const int* in_sizes, int n_in,
                              void* d_out, int out_size, void* d_ws, size_t ws_size,
                              hipStream_t stream) {
    const float* mask     = (const float*)d_in[0];
    const int*   instance = (const int*)  d_in[1];
    const float* W1       = (const float*)d_in[2];
    const float* b1       = (const float*)d_in[3];
    const float* W2       = (const float*)d_in[4];
    const float* b2       = (const float*)d_in[5];
    float*       out      = (float*)d_out;

    char* ws     = (char*)d_ws;
    int*  counts = (int*)ws;
    int*  lists  = (int*)(ws + LISTS_OFF);

    zero_counts_kernel<<<dim3(1), dim3(32), 0, stream>>>(counts);
    bucket_kernel<<<dim3(4), dim3(256), 0, stream>>>(instance, counts, lists);

    if (ws_size >= WS_NEEDED) {
        v16bf* BW1 = (v16bf*)(ws + BW1_OFF);
        v16bf* BW2 = (v16bf*)(ws + BW2_OFF);
        int total1 = NE * KC1 * NT1 * 32;   // 114688
        int total2 = NE * KC2 * NT2 * 32;   // 524288
        swizzle_w_kernel<<<dim3((total1 + 255) / 256), dim3(256), 0, stream>>>(
            W1, BW1, ND, KC1, NT1, NH, total1);
        swizzle_w_kernel<<<dim3((total2 + 255) / 256), dim3(256), 0, stream>>>(
            W2, BW2, NH, KC2, NT2, NA, total2);
        moe_mlp_swz_kernel<<<dim3(NE * TPE), dim3(256), 0, stream>>>(
            mask, counts, lists, BW1, b1, BW2, b2, out);
    } else {
        moe_mlp_direct_kernel<<<dim3(NE * TPE), dim3(256), 0, stream>>>(
            mask, counts, lists, W1, b1, W2, b2, out);
    }
}
